// Net_64046552318449
// MI455X (gfx1250) — matrix operand order repaired and checked by reading, not compiled, not run on hardware
//
#include <hip/hip_runtime.h>
#include <hip/hip_bf16.h>

typedef __attribute__((ext_vector_type(16))) _Float16 v16h;
typedef __attribute__((ext_vector_type(8)))  _Float16 v8h;
typedef __attribute__((ext_vector_type(8)))  float    v8f;

#define SEQ  2048
#define BATCH 128
#define HID   256
#define NTHREADS 512   // 16 waves of 32

extern __shared__ _Float16 smem[];   // 2 * BATCH*HID halves (double-buffered h)

// Persistent single-workgroup recurrent scan.
// Wave w: ntp = w%8 owns N-tiles {2*ntp, 2*ntp+1}; mg = w/8 owns M-tiles {4*mg..4*mg+3}.
__global__ void __launch_bounds__(NTHREADS)
rnn_scan_wmma(const float* __restrict__ x,
              const float* __restrict__ h0,
              const float* __restrict__ W_ih,
              const float* __restrict__ W_hh,
              const float* __restrict__ W_hh_bias,
              const float* __restrict__ b_h,
              const int*   __restrict__ ctx,
              float* __restrict__ out_h)
{
    const int tid  = threadIdx.x;
    const int lane = tid & 31;
    const int wave = tid >> 5;
    const int ntp  = wave & 7;          // N-tile pair index
    const int mg   = wave >> 3;         // M-group (0/1)
    const int l16  = lane & 15;
    const int hiA  = (lane >> 4) << 3;  // 0 or 8  : A/C layout upper-half K/M offset
    const int hiB  = (lane >> 4) << 4;  // 0 or 16 : unused (B fragments use hiA)

    const float ctxf = (float)ctx[0];

    // ---- preload W_eff^T as B fragments into registers -----------------
    // B tile (nt, kc) is 32x16 f16: n = nt*16 + l16, k = kc*32 + hiA + (e & 7) + ((e >> 3) << 4)
    v16h Bfrag[2][8];
#pragma unroll
    for (int nti = 0; nti < 2; ++nti) {
        const int n = (2 * ntp + nti) * 16 + l16;
#pragma unroll
        for (int kc = 0; kc < 8; ++kc) {
            v16h bf;
#pragma unroll
            for (int e = 0; e < 16; ++e) {
                const int k = kc * 32 + hiA + (e & 7) + ((e >> 3) << 4);
                const float v = W_hh[n * HID + k] + ctxf * W_hh_bias[n * HID + k];
                bf[e] = (_Float16)v;
            }
            Bfrag[nti][kc] = bf;
        }
    }

    // per-lane input weight / bias for this wave's two N columns
    const int n0 = (2 * ntp)     * 16 + l16;
    const int n1 = (2 * ntp + 1) * 16 + l16;
    const float win0 = W_ih[n0];
    const float win1 = W_ih[n1];
    const float bh0  = b_h[n0];
    const float bh1  = b_h[n1];

    // ---- initialize h buffer 0 with (2*h0 - 1) in f16 -------------------
    for (int i = tid; i < BATCH * HID; i += NTHREADS)
        smem[i] = (_Float16)(2.0f * h0[i] - 1.0f);
    __syncthreads();

    // ---- sequential scan: one workgroup barrier per step ----------------
    for (int t = 0; t < SEQ; ++t) {
        const _Float16* Hcur = smem + ((t & 1) ? BATCH * HID : 0);
        _Float16*       Hnxt = smem + ((t & 1) ? 0 : BATCH * HID);

#pragma unroll
        for (int mt = 0; mt < 4; ++mt) {
            const int mtile = (mg * 4 + mt) * 16;
            const int mrow  = mtile + l16;

            auto loadA = [&](int kc) -> v16h {
                const _Float16* p = Hcur + mrow * HID + kc * 32 + hiA;
                const v8h lo = *(const v8h*)(p);       // K = base + 0..7
                const v8h hi = *(const v8h*)(p + 16);  // K = base + 16..23
                return __builtin_shufflevector(lo, hi,
                    0,1,2,3,4,5,6,7,8,9,10,11,12,13,14,15);
            };

            // accumulator init: x_t * w_in + b_h (C layout: M = r + hiA)
            v8f acc0, acc1;
#pragma unroll
            for (int r = 0; r < 8; ++r) {
                const float xv = x[t * BATCH + mtile + r + hiA];
                acc0[r] = xv * win0 + bh0;
                acc1[r] = xv * win1 + bh1;
            }

            // GEMM over K, 2-deep software pipeline on the A fragments so
            // the next ds_load pair is in flight while the WMMAs execute.
            v16h a_cur = loadA(0);
#pragma unroll
            for (int kc = 0; kc < 8; ++kc) {
                v16h a_nxt = a_cur;
                if (kc < 7) a_nxt = loadA(kc + 1);
                acc0 = __builtin_amdgcn_wmma_f32_16x16x32_f16(
                    false, a_cur, false, Bfrag[0][kc], (short)0, acc0, false, false);
                acc1 = __builtin_amdgcn_wmma_f32_16x16x32_f16(
                    false, a_cur, false, Bfrag[1][kc], (short)0, acc1, false, false);
                a_cur = a_nxt;
            }

            // sigmoid via exp + v_rcp (no IEEE divide expansion):
            //   e = exp(-pre); h = rcp(1+e); 2h-1 = (1-e)*rcp(1+e)
#pragma unroll
            for (int r = 0; r < 8; ++r) {
                const int m = mtile + r + hiA;
                const float e0 = __expf(-acc0[r]);
                const float e1 = __expf(-acc1[r]);
                const float h0v = __builtin_amdgcn_rcpf(1.0f + e0);
                const float h1v = __builtin_amdgcn_rcpf(1.0f + e1);
                out_h[(size_t)m * (SEQ * HID) + (size_t)t * HID + n0] = h0v;
                out_h[(size_t)m * (SEQ * HID) + (size_t)t * HID + n1] = h1v;
                Hnxt[m * HID + n0] = (_Float16)((1.0f - e0) * h0v);
                Hnxt[m * HID + n1] = (_Float16)((1.0f - e1) * h1v);
            }
        }
        __syncthreads();
    }
}

// y[b,t] = dot(out[b,t,:], W[0,:]) + b[0] ; one wave per (b,t), wave32 reduce.
__global__ void y_project(const float* __restrict__ out_h,
                          const float* __restrict__ W,
                          const float* __restrict__ b,
                          float* __restrict__ y)
{
    const int gid  = blockIdx.x * blockDim.x + threadIdx.x;
    const int wv   = gid >> 5;
    const int lane = gid & 31;
    if (wv >= BATCH * SEQ) return;
    const float* row = out_h + (size_t)wv * HID;
    float s = 0.0f;
#pragma unroll
    for (int i = 0; i < 8; ++i) {
        const int j = lane + i * 32;
        s += row[j] * W[j];
    }
#pragma unroll
    for (int off = 16; off > 0; off >>= 1)
        s += __shfl_down(s, off, 32);
    if (lane == 0) y[wv] = s + b[0];
}

extern "C" void kernel_launch(void* const* d_in, const int* in_sizes, int n_in,
                              void* d_out, int out_size, void* d_ws, size_t ws_size,
                              hipStream_t stream) {
    const float* x      = (const float*)d_in[0];
    const float* h0     = (const float*)d_in[1];
    const float* W_ih   = (const float*)d_in[2];
    const float* W_hh   = (const float*)d_in[3];
    const float* W_hh_b = (const float*)d_in[4];
    const float* b_h    = (const float*)d_in[5];
    const float* W      = (const float*)d_in[6];
    const float* b      = (const float*)d_in[7];
    const int*   ctx    = (const int*)d_in[8];

    float* y     = (float*)d_out;                 // 128*2048
    float* out_h = y + (size_t)BATCH * SEQ;       // 128*2048*256

    const size_t lds_bytes = (size_t)2 * BATCH * HID * sizeof(_Float16); // 128 KB
    rnn_scan_wmma<<<1, NTHREADS, lds_bytes, stream>>>(
        x, h0, W_ih, W_hh, W_hh_b, b_h, ctx, out_h);

    const int nthr = BATCH * SEQ * 32;            // one wave per (b,t)
    y_project<<<(nthr + 255) / 256, 256, 0, stream>>>(out_h, W, b, y);
}
